// SambaReadout_60722247631156
// MI455X (gfx1250) — matrix operand order repaired and checked by reading, not compiled, not run on hardware
//
#include <hip/hip_runtime.h>

#define L_  12
#define B_  2
#define S_  2048
#define D_  768
#define QK_ 128
#define H_  512
#define V_  32000

typedef __attribute__((ext_vector_type(16))) __bf16 v16bf;
typedef __attribute__((ext_vector_type(8)))  float  v8f;
typedef __attribute__((ext_vector_type(4)))  unsigned int u32x4;
typedef __attribute__((ext_vector_type(8)))  int i32x8;
typedef __attribute__((ext_vector_type(4)))  int i32x4;
typedef unsigned short u16;

#if defined(__AMDGCN__) && __has_builtin(__builtin_amdgcn_tensor_load_to_lds) && \
    __has_builtin(__builtin_amdgcn_s_wait_tensorcnt)
#define HAVE_TDM 1
#else
#define HAVE_TDM 0
#endif

// f32 -> bf16 (round-to-nearest-even), raw 16-bit pattern
__device__ __forceinline__ u16 f2bf(float x) {
    union { float f; unsigned u; } t; t.f = x;
    unsigned r = t.u + 0x7FFFu + ((t.u >> 16) & 1u);
    return (u16)(r >> 16);
}
__device__ __forceinline__ float bf2f(u16 h) {
    union { unsigned u; float f; } t; t.u = ((unsigned)h) << 16;
    return t.f;
}

#if HAVE_TDM
// One TDM descriptor: 2D tile nrows x row_elems of bf16, row stride in elems,
// LDS destination padded 16B after every 64B (=> ushort row stride 40).
__device__ __forceinline__ void tdm_load_2d(unsigned lds_off, unsigned long long gaddr,
                                            unsigned row_elems, unsigned nrows,
                                            unsigned long long row_stride_elems)
{
    u32x4 g0;
    g0.x = 1u;                                          // count=1, user mode
    g0.y = lds_off;                                     // LDS byte address
    g0.z = (unsigned)(gaddr & 0xFFFFFFFFull);           // global_addr[31:0]
    g0.w = (unsigned)((gaddr >> 32) & 0x01FFFFFFull) | (2u << 30);  // [56:32] | type=2

    i32x8 g1;
    // data_size=2B (code 1), pad_enable, pad_interval=16 DW (code 3), pad_amount=4 DW (code 3)
    g1[0] = (1 << 16) | (1 << 20) | (3 << 22) | (3 << 25);
    g1[1] = (int)((row_elems & 0xFFFFu) << 16);                         // tensor_dim0[15:0]
    g1[2] = (int)(((row_elems >> 16) & 0xFFFFu) | ((nrows & 0xFFFFu) << 16)); // dim0 hi | dim1 lo
    g1[3] = (int)(((nrows >> 16) & 0xFFFFu) | ((row_elems & 0xFFFFu) << 16)); // dim1 hi | tile_dim0
    g1[4] = (int)(nrows & 0xFFFFu);                                     // tile_dim1 | tile_dim2=0
    g1[5] = (int)(row_stride_elems & 0xFFFFFFFFull);                    // dim0_stride lo
    g1[6] = (int)((row_stride_elems >> 32) & 0xFFFFull);                // dim0_stride hi | dim1_stride lo=0
    g1[7] = 0;

    i32x4 g2 = {0, 0, 0, 0};
    i32x4 g3 = {0, 0, 0, 0};
#if __clang_major__ >= 23
    i32x8 gx = {};
    __builtin_amdgcn_tensor_load_to_lds(g0, g1, g2, g3, gx, 0);
#else
    __builtin_amdgcn_tensor_load_to_lds(g0, g1, g2, g3, 0);
#endif
}
#endif

// ---------------------------------------------------------------------------
// bf16 WMMA GEMM: C[M,N] = act(A[M,K] * Bt[N,K]^T + bias)
// A, Bt bf16; C f32 or bf16 (flags bit1); relu (flags bit0).
// 256 threads = 8 wave32, 128x128 tile, wave => 32x64 (8 WMMA tiles), BK=32.
// ---------------------------------------------------------------------------
__global__ __launch_bounds__(256)
void wmma_gemm_bf16(const u16* __restrict__ A, const u16* __restrict__ Bt,
                    const float* __restrict__ bias,
                    void* __restrict__ C, int ldc, int K, int flags)
{
    // +8 u16 pad => row stride 80B: conflict-free 16B fragment reads
    __shared__ u16 As[2][128][40];   // [m][k]
    __shared__ u16 Bs[2][128][40];   // [n][k]

    const int tid  = threadIdx.x;
    const int lane = tid & 31;
    const int wv   = tid >> 5;
    const int wm   = (wv >> 1) * 32;     // wave M offset (0/32/64/96)
    const int wn   = (wv & 1) * 64;      // wave N offset (0/64)
    const int g    = lane >> 4;
    const int ln   = lane & 15;
    const int m0   = blockIdx.y * 128;
    const int n0   = blockIdx.x * 128;

    v8f acc[2][4] = {};
    int buf = 0;

#if HAVE_TDM
    if (wv == 0) {
        tdm_load_2d((unsigned)(size_t)&As[0][0][0],
                    (unsigned long long)(size_t)(A + (size_t)m0 * K), 32u, 128u,
                    (unsigned long long)K);
        tdm_load_2d((unsigned)(size_t)&Bs[0][0][0],
                    (unsigned long long)(size_t)(Bt + (size_t)n0 * K), 32u, 128u,
                    (unsigned long long)K);
    }
#endif

    for (int kk = 0; kk < K; kk += 32) {
#if HAVE_TDM
        if (wv == 0) __builtin_amdgcn_s_wait_tensorcnt(0);
        __syncthreads();
        asm volatile("" ::: "memory");
        if (kk + 32 < K && wv == 0) {   // prefetch next stage into other buffer
            tdm_load_2d((unsigned)(size_t)&As[buf ^ 1][0][0],
                        (unsigned long long)(size_t)(A + (size_t)m0 * K + kk + 32), 32u, 128u,
                        (unsigned long long)K);
            tdm_load_2d((unsigned)(size_t)&Bs[buf ^ 1][0][0],
                        (unsigned long long)(size_t)(Bt + (size_t)n0 * K + kk + 32), 32u, 128u,
                        (unsigned long long)K);
        }
#else
        #pragma unroll
        for (int i = 0; i < 2; ++i) {
            int c = tid + i * 256;                 // 512 chunks of 16B
            int row = c >> 2, cc = (c & 3) * 8;
            *(uint4*)&As[0][row][cc] = *(const uint4*)(A  + (size_t)(m0 + row) * K + kk + cc);
            *(uint4*)&Bs[0][row][cc] = *(const uint4*)(Bt + (size_t)(n0 + row) * K + kk + cc);
        }
        __syncthreads();
#endif

        // Fragments per ISA 16-bit layouts (two 16B LDS reads each)
        v16bf af[2], bfr[4];
        #pragma unroll
        for (int mt = 0; mt < 2; ++mt) {
            const u16* src = &As[buf][wm + mt * 16 + ln][0];
            ((uint4*)&af[mt])[0] = *(const uint4*)(src + 8 * g);       // K = 8g..8g+7
            ((uint4*)&af[mt])[1] = *(const uint4*)(src + 16 + 8 * g);  // K = 16+8g..
        }
        #pragma unroll
        for (int nt = 0; nt < 4; ++nt) {
            const u16* src = &Bs[buf][wn + nt * 16 + ln][0];
            ((uint4*)&bfr[nt])[0] = *(const uint4*)(src + 8 * g);      // K = 8g..8g+7
            ((uint4*)&bfr[nt])[1] = *(const uint4*)(src + 16 + 8 * g); // K = 16+8g..
        }

        #pragma unroll
        for (int mt = 0; mt < 2; ++mt)
            #pragma unroll
            for (int nt = 0; nt < 4; ++nt)
                acc[mt][nt] = __builtin_amdgcn_wmma_f32_16x16x32_bf16(
                    false, af[mt], false, bfr[nt], (short)0, acc[mt][nt], false, false);

        __syncthreads();
#if HAVE_TDM
        buf ^= 1;
#endif
    }

    const int relu = flags & 1, out16 = flags & 2;
    #pragma unroll
    for (int mt = 0; mt < 2; ++mt) {
        #pragma unroll
        for (int nt = 0; nt < 4; ++nt) {
            const int col = n0 + wn + nt * 16 + ln;
            const float bv = bias[col];
            #pragma unroll
            for (int j = 0; j < 8; ++j) {
                const int row = m0 + wm + mt * 16 + 8 * g + j;
                float x = acc[mt][nt][j] + bv;
                if (relu) x = fmaxf(x, 0.0f);
                const size_t idx = (size_t)row * ldc + col;
                if (out16) ((u16*)C)[idx] = f2bf(x);
                else       ((float*)C)[idx] = x;
            }
        }
    }
}

// ---------------------------------------------------------------------------
// Pre-conversion passes
// ---------------------------------------------------------------------------
__global__ __launch_bounds__(256)
void cvt_bf16_vec(const float* __restrict__ src, u16* __restrict__ dst)
{
    const int i = blockIdx.x * 256 + threadIdx.x;   // float4 index
    const float4 v = ((const float4*)src)[i];
    ushort4 o;
    o.x = f2bf(v.x); o.y = f2bf(v.y); o.z = f2bf(v.z); o.w = f2bf(v.w);
    ((ushort4*)dst)[i] = o;
}

__global__ __launch_bounds__(256)
void cvt_w_transpose(const float* __restrict__ W, u16* __restrict__ Wt, int Kd, int Nd)
{
    const int i = blockIdx.x * 256 + threadIdx.x;   // over Kd*Nd
    if (i >= Kd * Nd) return;
    const int k = i / Nd, n = i % Nd;
    Wt[(size_t)n * Kd + k] = f2bf(W[i]);
}

// q_in[s*B+b, d] = bf16(mean over l of X[l,b,s,d])
__global__ __launch_bounds__(256)
void mean_layers(const float* __restrict__ x, u16* __restrict__ q_in)
{
    const int i = blockIdx.x * 256 + threadIdx.x;   // over S*B*D
    const int d = i % D_;
    const int r = i / D_;                           // s*B + b
    const int b = r % B_;
    const int s = r / B_;
    float acc = 0.f;
    #pragma unroll
    for (int l = 0; l < L_; ++l)
        acc += x[(size_t)((l * B_ + b) * S_ + s) * D_ + d];
    q_in[(size_t)r * D_ + d] = f2bf(acc * (1.0f / L_));
}

// ---------------------------------------------------------------------------
// Per (s,b): scores over L, softmax, weighted v sum -> agg[b,s,h] (bf16)
// ---------------------------------------------------------------------------
__global__ __launch_bounds__(128)
void attn_combine(const u16* __restrict__ q, const u16* __restrict__ k,
                  const u16* __restrict__ v, u16* __restrict__ agg)
{
    __shared__ float sc[L_];
    const int tid  = threadIdx.x;
    const int lane = tid & 31;
    const int w    = tid >> 5;
    const int r2   = blockIdx.x;        // s*B + b
    const int b    = r2 % B_;
    const int s    = r2 / B_;
    const float scale = 0.08838834764831845f;   // 128^-0.5

    for (int l = w; l < L_; l += 4) {
        const u16* qp = q + (size_t)r2 * QK_;
        const u16* kp = k + (size_t)((l * B_ + b) * S_ + s) * QK_;
        float p = 0.f;
        #pragma unroll
        for (int e = 0; e < 4; ++e)
            p += bf2f(qp[lane + 32 * e]) * bf2f(kp[lane + 32 * e]);
        #pragma unroll
        for (int off = 16; off > 0; off >>= 1)   // wave32 reduction
            p += __shfl_xor(p, off, 32);
        if (lane == 0) sc[l] = p * scale;
    }
    __syncthreads();

    float ex[L_];
    float mx = -1e30f;
    #pragma unroll
    for (int l = 0; l < L_; ++l) mx = fmaxf(mx, sc[l]);
    float sum = 0.f;
    #pragma unroll
    for (int l = 0; l < L_; ++l) { ex[l] = __expf(sc[l] - mx); sum += ex[l]; }
    const float inv = 1.0f / sum;

    for (int h = tid; h < H_; h += 128) {
        float acc = 0.f;
        #pragma unroll
        for (int l = 0; l < L_; ++l)
            acc += ex[l] * bf2f(v[(size_t)((l * B_ + b) * S_ + s) * H_ + h]);
        agg[(size_t)(b * S_ + s) * H_ + h] = f2bf(acc * inv);   // (B,S,H)
    }
}

// ---------------------------------------------------------------------------
extern "C" void kernel_launch(void* const* d_in, const int* in_sizes, int n_in,
                              void* d_out, int out_size, void* d_ws, size_t ws_size,
                              hipStream_t stream)
{
    const float* X  = (const float*)d_in[0];
    const float* Wq = (const float*)d_in[1];
    const float* bq = (const float*)d_in[2];
    const float* Wk = (const float*)d_in[3];
    const float* bk = (const float*)d_in[4];
    const float* Wv = (const float*)d_in[5];
    const float* bv = (const float*)d_in[6];
    const float* W1 = (const float*)d_in[7];
    const float* b1 = (const float*)d_in[8];
    const float* W2 = (const float*)d_in[9];
    const float* b2 = (const float*)d_in[10];

    // workspace layout (u16 units, all offsets 16B-aligned)
    u16* ws   = (u16*)d_ws;
    u16* xbf  = ws;                                   // L*B*S*D
    u16* wqt  = xbf  + (size_t)L_ * B_ * S_ * D_;     // QK x D
    u16* wkt  = wqt  + (size_t)QK_ * D_;              // QK x D
    u16* wvt  = wkt  + (size_t)QK_ * D_;              // H  x D
    u16* w1t  = wvt  + (size_t)H_ * D_;               // (H/2) x H
    u16* w2t  = w1t  + (size_t)(H_ / 2) * H_;         // V x (H/2)
    u16* qin  = w2t  + (size_t)V_ * (H_ / 2);         // (S*B) x D
    u16* qb   = qin  + (size_t)S_ * B_ * D_;          // (S*B) x QK
    u16* kb   = qb   + (size_t)S_ * B_ * QK_;         // (L*B*S) x QK
    u16* vb   = kb   + (size_t)L_ * B_ * S_ * QK_;    // (L*B*S) x H
    u16* aggb = vb   + (size_t)L_ * B_ * S_ * H_;     // (B*S) x H
    u16* midb = aggb + (size_t)B_ * S_ * H_;          // (B*S) x (H/2)

    // 0) one-time conversions: X -> bf16, weights -> bf16 transposed [N][K]
    cvt_bf16_vec<<<(L_ * B_ * S_ * D_) / 4 / 256, 256, 0, stream>>>(X, xbf);
    cvt_w_transpose<<<(D_ * QK_ + 255) / 256, 256, 0, stream>>>(Wq, wqt, D_, QK_);
    cvt_w_transpose<<<(D_ * QK_ + 255) / 256, 256, 0, stream>>>(Wk, wkt, D_, QK_);
    cvt_w_transpose<<<(D_ * H_ + 255) / 256, 256, 0, stream>>>(Wv, wvt, D_, H_);
    cvt_w_transpose<<<(H_ * (H_ / 2) + 255) / 256, 256, 0, stream>>>(W1, w1t, H_, H_ / 2);
    cvt_w_transpose<<<((H_ / 2) * V_ + 255) / 256, 256, 0, stream>>>(W2, w2t, H_ / 2, V_);
    // 1) mean over layers -> q_in (bf16)
    mean_layers<<<(S_ * B_ * D_) / 256, 256, 0, stream>>>(X, qin);
    // 2) q = relu(q_in @ Wq + bq)          M=4096  K=768 N=128 -> bf16
    wmma_gemm_bf16<<<dim3(1, 32), 256, 0, stream>>>(qin, wqt, bq, qb, QK_, D_, 3);
    // 3) k = relu(X @ Wk + bk)             M=49152 K=768 N=128 -> bf16
    wmma_gemm_bf16<<<dim3(1, 384), 256, 0, stream>>>(xbf, wkt, bk, kb, QK_, D_, 3);
    // 4) v = X @ Wv + bv                   M=49152 K=768 N=512 -> bf16
    wmma_gemm_bf16<<<dim3(4, 384), 256, 0, stream>>>(xbf, wvt, bv, vb, H_, D_, 2);
    // 5) softmax over L + weighted v sum -> agg (B,S,H) bf16
    attn_combine<<<S_ * B_, 128, 0, stream>>>(qb, kb, vb, aggb);
    // 6) mid = relu(agg @ W1 + b1)         M=4096  K=512 N=256 -> bf16
    wmma_gemm_bf16<<<dim3(2, 32), 256, 0, stream>>>(aggb, w1t, b1, midb, H_ / 2, H_, 3);
    // 7) out = mid @ W2 + b2               M=4096  K=256 N=32000 -> f32
    wmma_gemm_bf16<<<dim3(250, 32), 256, 0, stream>>>(midb, w2t, b2, d_out, V_, H_ / 2, 0);
}
